// PerceiverAttention_46368466927991
// MI455X (gfx1250) — compile-verified, matrix operand order checked by reading, not run on hardware
//
#include <hip/hip_runtime.h>
#include <hip/hip_bf16.h>

typedef __attribute__((ext_vector_type(16))) __bf16 v16bf;
typedef __attribute__((ext_vector_type(8)))  __bf16 v8bf;
typedef __attribute__((ext_vector_type(8)))  float  v8f;
typedef __attribute__((ext_vector_type(4)))  unsigned int v4u;
typedef __attribute__((ext_vector_type(8)))  int v8i;
typedef __attribute__((ext_vector_type(4)))  int v4i;

#define B_      8
#define MSEQ    4096
#define NLAT    64
#define DIM_    1024
#define HEADS_  16
#define DHEAD_  64
#define INNER_  1024

// ---------------------------------------------------------------------------
// WMMA helpers (CDNA5 gfx1250, wave32)
// ---------------------------------------------------------------------------
__device__ __forceinline__ v8f wmma_bf16(v16bf a, v16bf b, v8f c) {
  return __builtin_amdgcn_wmma_f32_16x16x32_bf16(false, a, false, b, (short)0, c,
                                                 false, false);
}

// 16-bit A/B fragment from a K-contiguous LDS row (ISA 7.12.2):
// lane L (half=L/16) holds K = kbase + (e/8)*16 + half*8 + (e%8)
__device__ __forceinline__ v16bf load_frag(const __bf16* row_ptr, int kbase, int half) {
  const __bf16* p0 = row_ptr + kbase + half * 8;
  v8bf lo = *(const v8bf*)(p0);
  v8bf hi = *(const v8bf*)(p0 + 16);
  v16bf r;
#pragma unroll
  for (int i = 0; i < 8; ++i) { r[i] = lo[i]; r[i + 8] = hi[i]; }
  return r;
}

// ---------------------------------------------------------------------------
// Tensor Data Mover: async 2D bf16 tile load Global -> LDS (D# per ISA ch8).
// tile_w elements per row, tile_h rows, row stride in elements.
// ---------------------------------------------------------------------------
__device__ __forceinline__ unsigned lds_addr_of(const void* p) {
  // generic (flat) LDS pointer: low 32 bits are the wave-relative LDS address
  return (unsigned)(unsigned long long)p;
}

__device__ __forceinline__ void tdm_load_2d_bf16(const __bf16* gsrc, unsigned lds_byte,
                                                 unsigned tile_w, unsigned tile_h,
                                                 unsigned row_stride_elems) {
  unsigned long long ga = (unsigned long long)gsrc;
  v4u g0;
  g0[0] = 1u;                                                  // count=1 (valid D#)
  g0[1] = lds_byte;                                            // lds_addr
  g0[2] = (unsigned)ga;                                        // global_addr[31:0]
  g0[3] = (unsigned)((ga >> 32) & 0x01ffffffu) | (2u << 30);   // addr[56:32]|type=2

  const unsigned td0 = 1u << 20, td1 = 1u << 20;  // generous tensor dims (no OOB)
  v8i g1;
  g1[0] = (int)(1u << 16);                                     // data_size=1 (2B)
  g1[1] = (int)((td0 & 0xffffu) << 16);                        // tensor_dim0 lo
  g1[2] = (int)((td0 >> 16) | ((td1 & 0xffffu) << 16));        // td0 hi | td1 lo
  g1[3] = (int)((td1 >> 16) | (tile_w << 16));                 // td1 hi | tile_dim0
  g1[4] = (int)(tile_h & 0xffffu);                             // tile_dim1 (dim2=0)
  g1[5] = (int)row_stride_elems;                               // tensor_dim0_stride
  g1[6] = 0;
  g1[7] = 0;

  v4i z4; z4[0] = 0; z4[1] = 0; z4[2] = 0; z4[3] = 0;
  v8i z8;
#pragma unroll
  for (int i = 0; i < 8; ++i) z8[i] = 0;
  // amdgpu-toolchain (clang-23) 6-arg form
  __builtin_amdgcn_tensor_load_to_lds(g0, g1, z4, z4, z8, 0);
}

// ---------------------------------------------------------------------------
// f32 -> bf16 transposing convert for weights: src[K][N] -> dst[N][K]
// ---------------------------------------------------------------------------
__global__ __launch_bounds__(256) void cvt_transpose_bf16(const float* __restrict__ src,
                                                          __bf16* __restrict__ dst,
                                                          int K, int N) {
  const int n = blockIdx.x * 32 + (threadIdx.x & 31);
  const int k = blockIdx.y * 8 + (threadIdx.x >> 5);
  if (n < N && k < K) dst[(size_t)n * K + k] = (__bf16)src[(size_t)k * N + n];
}

// ---------------------------------------------------------------------------
// LayerNorm over last dim (1024) -> bf16.  One 256-thread block per row.
// ---------------------------------------------------------------------------
__global__ __launch_bounds__(256) void ln_bf16_kernel(const float* __restrict__ x,
                                                      const float* __restrict__ gam,
                                                      const float* __restrict__ bet,
                                                      __bf16* __restrict__ out) {
  __shared__ float red[256];
  const int row = blockIdx.x;
  const int tid = threadIdx.x;
  const float* xr = x + (size_t)row * DIM_;

  float4 xv = *(const float4*)(xr + tid * 4);
  red[tid] = xv.x + xv.y + xv.z + xv.w;
  __syncthreads();
#pragma unroll
  for (int off = 128; off >= 1; off >>= 1) {
    if (tid < off) red[tid] += red[tid + off];
    __syncthreads();
  }
  const float mu = red[0] * (1.0f / DIM_);
  __syncthreads();

  float dx = xv.x - mu, dy = xv.y - mu, dz = xv.z - mu, dw = xv.w - mu;
  red[tid] = dx * dx + dy * dy + dz * dz + dw * dw;
  __syncthreads();
#pragma unroll
  for (int off = 128; off >= 1; off >>= 1) {
    if (tid < off) red[tid] += red[tid + off];
    __syncthreads();
  }
  const float rstd = rsqrtf(red[0] * (1.0f / DIM_) + 1e-5f);

  float4 gv = *(const float4*)(gam + tid * 4);
  float4 bv = *(const float4*)(bet + tid * 4);
  __bf16* o = out + (size_t)row * DIM_ + tid * 4;
  o[0] = (__bf16)(dx * rstd * gv.x + bv.x);
  o[1] = (__bf16)(dy * rstd * gv.y + bv.y);
  o[2] = (__bf16)(dz * rstd * gv.z + bv.z);
  o[3] = (__bf16)(dw * rstd * gv.w + bv.w);
}

// ---------------------------------------------------------------------------
// bf16 GEMM: C[M,N] = A[M,1024] @ B, with B given pre-transposed BT[N][1024].
// 128x128 tile / 256 threads (8 waves, each 32x64 = 2x4 WMMA tiles), K-step 64
// (16 WMMA per stage).  Tiles staged by TDM with 2-deep double buffering:
// wave 0 prefetches tile k+1 while all waves compute tile k; since TENSORcnt
// retires in order, s_wait_tensorcnt(2) == "tile k's pair has landed".
//   MODE 0: N=2048 -> K as bf16 [b,h,m,d]; V as bf16 [b,h,d,m] (pre-transposed)
//   MODE 1: N=1024 -> q as bf16 [b,h,n,d]
//   MODE 2: N=1024 -> f32 row-major outF
// ---------------------------------------------------------------------------
template <int MODE>
__global__ __launch_bounds__(256) void gemm_bf16_kernel(
    const __bf16* __restrict__ A, const __bf16* __restrict__ BT, int M, int N,
    __bf16* __restrict__ outK, __bf16* __restrict__ outV, float* __restrict__ outF) {
  const int K = DIM_;
  const int NKT = K / 64;  // 16 K-tiles
  __shared__ __align__(16) __bf16 a_lds[2][128 * 64];   // [buf][row][k]
  __shared__ __align__(16) __bf16 bT_lds[2][128 * 64];  // [buf][col][k]

  const int tid  = threadIdx.x;
  const int lane = tid & 31;
  const int wid  = tid >> 5;
  const int half = (lane >> 4) & 1;
  const int l16  = lane & 15;
  const int m0 = blockIdx.y * 128;
  const int n0 = blockIdx.x * 128;
  const int rowbase = (wid & 3) * 32;
  const int colbase = (wid >> 2) * 64;

  v8f acc[2][4];
#pragma unroll
  for (int r = 0; r < 2; ++r)
#pragma unroll
    for (int c = 0; c < 4; ++c)
#pragma unroll
      for (int i = 0; i < 8; ++i) acc[r][c][i] = 0.0f;

  const __bf16* agp = A + (size_t)m0 * K;
  const __bf16* bgp = BT + (size_t)n0 * K;

  if (tid < 32) {  // prologue: prefetch tile 0 into buffer 0
    tdm_load_2d_bf16(agp, lds_addr_of(a_lds[0]), 64, 128, K);
    tdm_load_2d_bf16(bgp, lds_addr_of(bT_lds[0]), 64, 128, K);
  }

  for (int kt = 0; kt < NKT; ++kt) {
    const int cur = kt & 1, nxt = cur ^ 1;
    __syncthreads();  // consumers of buf[nxt] (iteration kt-1) are done
    if (tid < 32) {
      if (kt + 1 < NKT) {  // prefetch tile kt+1, then wait for tile kt only
        tdm_load_2d_bf16(agp + (kt + 1) * 64, lds_addr_of(a_lds[nxt]), 64, 128, K);
        tdm_load_2d_bf16(bgp + (kt + 1) * 64, lds_addr_of(bT_lds[nxt]), 64, 128, K);
        __builtin_amdgcn_s_wait_tensorcnt(2);
      } else {
        __builtin_amdgcn_s_wait_tensorcnt(0);
      }
    }
    __syncthreads();  // tile kt visible to all waves

    v16bf af[2][2], bf[4][2];
#pragma unroll
    for (int r = 0; r < 2; ++r)
#pragma unroll
      for (int kf = 0; kf < 2; ++kf)
        af[r][kf] =
            load_frag(a_lds[cur] + (rowbase + r * 16 + l16) * 64, kf * 32, half);
#pragma unroll
    for (int c = 0; c < 4; ++c)
#pragma unroll
      for (int kf = 0; kf < 2; ++kf)
        bf[c][kf] =
            load_frag(bT_lds[cur] + (colbase + c * 16 + l16) * 64, kf * 32, half);
#pragma unroll
    for (int kf = 0; kf < 2; ++kf)
#pragma unroll
      for (int r = 0; r < 2; ++r)
#pragma unroll
        for (int c = 0; c < 4; ++c)
          acc[r][c] = wmma_bf16(af[r][kf], bf[c][kf], acc[r][c]);
  }

  // C/D layout: row = elem + 8*half, col = lane%16
#pragma unroll
  for (int r = 0; r < 2; ++r) {
#pragma unroll
    for (int c = 0; c < 4; ++c) {
#pragma unroll
      for (int i = 0; i < 8; ++i) {
        const int gr = m0 + rowbase + r * 16 + i + half * 8;
        const int gc = n0 + colbase + c * 16 + l16;
        const float val = acc[r][c][i];
        if (MODE == 0) {
          const int bb = gr >> 12, mm = gr & 4095;
          if (gc < INNER_) {
            const int h = gc >> 6, d = gc & 63;
            outK[((size_t)(bb * HEADS_ + h) * MSEQ + mm) * DHEAD_ + d] = (__bf16)val;
          } else {
            const int c2 = gc - INNER_;
            const int h = c2 >> 6, d = c2 & 63;
            outV[((size_t)(bb * HEADS_ + h) * DHEAD_ + d) * MSEQ + mm] = (__bf16)val;
          }
        } else if (MODE == 1) {
          const int bb = gr >> 6, nn = gr & 63;
          const int h = gc >> 6, d = gc & 63;
          outK[((size_t)(bb * HEADS_ + h) * NLAT + nn) * DHEAD_ + d] = (__bf16)val;
        } else {
          outF[(size_t)gr * N + gc] = val;
        }
      }
    }
  }
}

// ---------------------------------------------------------------------------
// Flash attention per (b,h): 128 threads = 4 waves, wave w owns q rows
// [16w,16w+16), m in 64-wide tiles with online softmax.  K/V tiles streamed
// via TDM with double buffering (prefetch mt+1 while computing mt).
// q bf16 [b,h,n,64]; K bf16 [b,h,m,64]; V bf16 [b,h,64,m] (pre-transposed);
// out bf16 [b, n, h*64+d] = [512,1024] row-major.
// ---------------------------------------------------------------------------
__global__ __launch_bounds__(128) void attn_kernel(
    const __bf16* __restrict__ Q, const __bf16* __restrict__ Km,
    const __bf16* __restrict__ Vt, __bf16* __restrict__ Ob) {
  __shared__ __align__(16) __bf16 q_lds[64 * 64];
  __shared__ __align__(16) __bf16 k_lds[2][64 * 64];   // [buf][m][d]
  __shared__ __align__(16) __bf16 vT_lds[2][64 * 64];  // [buf][d][m]
  __shared__ __align__(16) __bf16 p_lds[4 * 16 * 64];  // per-wave P [16][64]

  const int bh = blockIdx.x;
  const int b = bh >> 4, h = bh & 15;
  const int tid  = threadIdx.x;
  const int lane = tid & 31;
  const int wave = tid >> 5;
  const int half = (lane >> 4) & 1;
  const int l16  = lane & 15;

  const __bf16* qg  = Q  + (size_t)bh * NLAT * DHEAD_;
  const __bf16* kg  = Km + (size_t)bh * MSEQ * DHEAD_;
  const __bf16* vgT = Vt + (size_t)bh * DHEAD_ * MSEQ;  // [64][4096]

  if (tid < 32) {  // prologue: prefetch m-tile 0 into buffer 0
    tdm_load_2d_bf16(kg, lds_addr_of(k_lds[0]), 64, 64, DHEAD_);
    tdm_load_2d_bf16(vgT, lds_addr_of(vT_lds[0]), 64, 64, MSEQ);
  }

  {  // stage q 64x64 (once)
    const int row = tid >> 1, c0 = (tid & 1) * 32;
#pragma unroll
    for (int i = 0; i < 4; ++i)
      *(v8bf*)(q_lds + row * 64 + c0 + i * 8) =
          *(const v8bf*)(qg + (size_t)row * 64 + c0 + i * 8);
  }
  __syncthreads();

  v16bf aq[2];
  aq[0] = load_frag(q_lds + (wave * 16 + l16) * 64, 0, half);
  aq[1] = load_frag(q_lds + (wave * 16 + l16) * 64, 32, half);

  v8f Oacc[4];
  float m_i[8], l_i[8];
#pragma unroll
  for (int c = 0; c < 4; ++c)
#pragma unroll
    for (int i = 0; i < 8; ++i) Oacc[c][i] = 0.0f;
#pragma unroll
  for (int i = 0; i < 8; ++i) { m_i[i] = -1e30f; l_i[i] = 0.0f; }

  const float scl = 0.125f;  // DIM_HEAD^-0.5
  __bf16* pw = p_lds + wave * 16 * 64;
  const int NMT = MSEQ / 64;

  for (int mt = 0; mt < NMT; ++mt) {
    const int cur = mt & 1, nxt = cur ^ 1;
    __syncthreads();  // consumers of buf[nxt] (iteration mt-1) are done
    if (tid < 32) {
      if (mt + 1 < NMT) {  // prefetch mt+1, wait only for mt's pair
        tdm_load_2d_bf16(kg + (size_t)(mt + 1) * 64 * DHEAD_,
                         lds_addr_of(k_lds[nxt]), 64, 64, DHEAD_);
        tdm_load_2d_bf16(vgT + (mt + 1) * 64, lds_addr_of(vT_lds[nxt]), 64, 64, MSEQ);
        __builtin_amdgcn_s_wait_tensorcnt(2);
      } else {
        __builtin_amdgcn_s_wait_tensorcnt(0);
      }
    }
    __syncthreads();

    // sim = scale * Q K^T : 4 tiles of 16x16, K=64
    v8f sim[4];
#pragma unroll
    for (int ct = 0; ct < 4; ++ct) {
      v8f s;
#pragma unroll
      for (int i = 0; i < 8; ++i) s[i] = 0.0f;
      const __bf16* krow = k_lds[cur] + (ct * 16 + l16) * 64;
      s = wmma_bf16(aq[0], load_frag(krow, 0, half), s);
      s = wmma_bf16(aq[1], load_frag(krow, 32, half), s);
#pragma unroll
      for (int i = 0; i < 8; ++i) s[i] *= scl;
      sim[ct] = s;
    }

    // online softmax: row r = i + 8*half lives in 16 lanes sharing bit4
    float corr[8];
#pragma unroll
    for (int i = 0; i < 8; ++i) {
      float mx = -1e30f;
#pragma unroll
      for (int ct = 0; ct < 4; ++ct) mx = fmaxf(mx, sim[ct][i]);
#pragma unroll
      for (int off = 1; off < 16; off <<= 1) mx = fmaxf(mx, __shfl_xor(mx, off, 32));
      const float mnew = fmaxf(m_i[i], mx);
      const float cr = __expf(m_i[i] - mnew);
      m_i[i] = mnew;
      float rs = 0.0f;
#pragma unroll
      for (int ct = 0; ct < 4; ++ct) {
        const float p = __expf(sim[ct][i] - mnew);
        sim[ct][i] = p;
        rs += p;
      }
#pragma unroll
      for (int off = 1; off < 16; off <<= 1) rs += __shfl_xor(rs, off, 32);
      l_i[i] = l_i[i] * cr + rs;
      corr[i] = cr;
    }
#pragma unroll
    for (int c = 0; c < 4; ++c)
#pragma unroll
      for (int i = 0; i < 8; ++i) Oacc[c][i] *= corr[i];

    // P (C/D layout) -> wave-private LDS [16][64] for A fragments
#pragma unroll
    for (int ct = 0; ct < 4; ++ct)
#pragma unroll
      for (int i = 0; i < 8; ++i)
        pw[(i + half * 8) * 64 + ct * 16 + l16] = (__bf16)sim[ct][i];

    // O += P V
    const v16bf ap0 = load_frag(pw + l16 * 64, 0, half);
    const v16bf ap1 = load_frag(pw + l16 * 64, 32, half);
#pragma unroll
    for (int dc = 0; dc < 4; ++dc) {
      const __bf16* vrow = vT_lds[cur] + (dc * 16 + l16) * 64;
      Oacc[dc] = wmma_bf16(ap0, load_frag(vrow, 0, half), Oacc[dc]);
      Oacc[dc] = wmma_bf16(ap1, load_frag(vrow, 32, half), Oacc[dc]);
    }
  }

  // normalize and store as [b, n, h*64+d]
#pragma unroll
  for (int dc = 0; dc < 4; ++dc)
#pragma unroll
    for (int i = 0; i < 8; ++i) {
      const int nrow = wave * 16 + i + half * 8;
      const int d = dc * 16 + l16;
      const float val = Oacc[dc][i] / l_i[i];
      Ob[(size_t)(b * NLAT + nrow) * INNER_ + h * DHEAD_ + d] = (__bf16)val;
    }
}

// ---------------------------------------------------------------------------
// Launch
// ---------------------------------------------------------------------------
extern "C" void kernel_launch(void* const* d_in, const int* in_sizes, int n_in,
                              void* d_out, int out_size, void* d_ws, size_t ws_size,
                              hipStream_t stream) {
  const float* x    = (const float*)d_in[0];
  const float* lat  = (const float*)d_in[1];
  const float* Wq   = (const float*)d_in[2];
  const float* Wkv  = (const float*)d_in[3];
  const float* Wout = (const float*)d_in[4];
  const float* gam  = (const float*)d_in[5];
  const float* bet  = (const float*)d_in[6];
  float* out = (float*)d_out;

  char* ws = (char*)d_ws;
  size_t off = 0;
  auto alloc = [&](size_t bytes) {
    char* p = ws + off;
    off += (bytes + 255) & ~(size_t)255;
    return p;
  };
  __bf16* WqT_bf   = (__bf16*)alloc((size_t)DIM_ * INNER_ * 2);      // [1024][1024]
  __bf16* WkvT_bf  = (__bf16*)alloc((size_t)2 * INNER_ * DIM_ * 2);  // [2048][1024]
  __bf16* WoutT_bf = (__bf16*)alloc((size_t)DIM_ * INNER_ * 2);      // [1024][1024]
  __bf16* xn_bf    = (__bf16*)alloc((size_t)B_ * MSEQ * DIM_ * 2);
  __bf16* lat_bf   = (__bf16*)alloc((size_t)B_ * NLAT * DIM_ * 2);
  __bf16* k_bf     = (__bf16*)alloc((size_t)B_ * HEADS_ * MSEQ * DHEAD_ * 2);
  __bf16* vT_bf    = (__bf16*)alloc((size_t)B_ * HEADS_ * DHEAD_ * MSEQ * 2);
  __bf16* q_bf     = (__bf16*)alloc((size_t)B_ * HEADS_ * NLAT * DHEAD_ * 2);
  __bf16* at_bf    = (__bf16*)alloc((size_t)B_ * NLAT * INNER_ * 2);

  // transpose + down-convert weights (one-time, tiny vs GEMM traffic)
  cvt_transpose_bf16<<<dim3(INNER_ / 32, DIM_ / 8), 256, 0, stream>>>(
      Wq, WqT_bf, DIM_, INNER_);
  cvt_transpose_bf16<<<dim3(2 * INNER_ / 32, DIM_ / 8), 256, 0, stream>>>(
      Wkv, WkvT_bf, DIM_, 2 * INNER_);
  cvt_transpose_bf16<<<dim3(DIM_ / 32, INNER_ / 8), 256, 0, stream>>>(
      Wout, WoutT_bf, INNER_, DIM_);

  ln_bf16_kernel<<<B_ * MSEQ, 256, 0, stream>>>(x, gam, bet, xn_bf);
  ln_bf16_kernel<<<B_ * NLAT, 256, 0, stream>>>(lat, gam, bet, lat_bf);

  // kv = xn @ Wkv  (M=32768, N=2048) -> K [b,h,m,d], V [b,h,d,m]
  gemm_bf16_kernel<0><<<dim3(16, 256), 256, 0, stream>>>(
      xn_bf, WkvT_bf, B_ * MSEQ, 2 * INNER_, k_bf, vT_bf, nullptr);
  // q = ln(latents) @ Wq  (M=512, N=1024) -> [b,h,n,d]
  gemm_bf16_kernel<1><<<dim3(8, 4), 256, 0, stream>>>(
      lat_bf, WqT_bf, B_ * NLAT, INNER_, q_bf, nullptr, nullptr);
  // attention per (b,h)
  attn_kernel<<<B_ * HEADS_, 128, 0, stream>>>(q_bf, k_bf, vT_bf, at_bf);
  // out = attn @ Wout  (M=512, N=1024, f32)
  gemm_bf16_kernel<2><<<dim3(8, 4), 256, 0, stream>>>(
      at_bf, WoutT_bf, B_ * NLAT, INNER_, nullptr, nullptr, out);
}
